// PointSAM_54906861912678
// MI455X (gfx1250) — matrix operand order repaired and checked by reading, not compiled, not run on hardware
//
#include <hip/hip_runtime.h>
#include <math.h>

typedef __attribute__((ext_vector_type(16))) _Float16 v16h;
typedef __attribute__((ext_vector_type(8)))  _Float16 v8h;
typedef __attribute__((ext_vector_type(8)))  float    v8f;

// Problem dims
#define BB 2
#define NN 8192
#define II 32
#define EE 128
#define HH 64

// Workspace byte offsets (all 256B aligned)
#define WS_Q      ((size_t)0)          // B*N*64 f32 = 4,194,304 B
#define WS_W2H    ((size_t)4194304)    // 128*64 f16 = 16384 B
#define WS_MH     ((size_t)4210688)    // 64*64 f16  = 8192 B
#define WS_W1EFF  ((size_t)4218880)    // 64*3 f32   (pad to 1024)
#define WS_WPD    ((size_t)4219904)    // 64*3 f32   (pad to 1024)
#define WS_BCONST ((size_t)4220928)    // 64 f32
#define WS_PINIT  ((size_t)4221184)    // 2*32*64 f32 = 16384 B
#define WS_PREF   ((size_t)4237568)    // 16384 B
#define WS_INTER  ((size_t)4253952)    // 2*32*32 u32 = 8192 B

// ---------------------------------------------------------------------------
// Prep: fold LoRA into effective weights, precompute fused small matrices.
//   W1eff[64][3]  = W1 + 0.25*B1@A1
//   W2eff[128][64]= W2 + 0.25*B2@A2            (f16 for WMMA B operand)
//   M[64][64]     = Wd1 @ W2eff                (f16; Q = h0 @ M.T fused GEMM)
//   Wpd[64][3]    = Wd1 @ Wp,  bconst = Wd1@bp + bd1
//   Pinit[b][i][c]= pos_coords @ Wpd.T + bconst
// ---------------------------------------------------------------------------
__global__ __launch_bounds__(256)
void prep_kernel(const float* __restrict__ W1, const float* __restrict__ A1,
                 const float* __restrict__ B1, const float* __restrict__ W2,
                 const float* __restrict__ A2, const float* __restrict__ B2,
                 const float* __restrict__ Wp, const float* __restrict__ bp,
                 const float* __restrict__ Wd1, const float* __restrict__ bd1,
                 const float* __restrict__ pos_coords,
                 float* __restrict__ W1eff, _Float16* __restrict__ W2h,
                 _Float16* __restrict__ Mh, float* __restrict__ WpdG,
                 float* __restrict__ bconstG, float* __restrict__ Pinit)
{
    __shared__ float sW2f[128 * 64];
    __shared__ float sWpd[64 * 3];
    __shared__ float sbc[64];
    const int t = threadIdx.x;

    if (t < 192) {  // W1eff
        int c = t / 3, d = t % 3;
        float v = W1[t];
        for (int r = 0; r < 4; ++r) v += 0.25f * B1[c * 4 + r] * A1[r * 3 + d];
        W1eff[t] = v;
    }
    for (int idx = t; idx < 128 * 64; idx += 256) {  // W2eff
        int n = idx >> 6, k = idx & 63;
        float v = W2[idx];
        for (int r = 0; r < 4; ++r) v += 0.25f * B2[n * 4 + r] * A2[r * 64 + k];
        sW2f[idx] = v;
        W2h[idx] = (_Float16)v;
    }
    if (t < 192) {  // Wpd
        int c = t / 3, d = t % 3;
        float v = 0.0f;
        for (int n = 0; n < 128; ++n) v += Wd1[c * 128 + n] * Wp[n * 3 + d];
        sWpd[t] = v;
        WpdG[t] = v;
    }
    if (t < 64) {   // bconst
        float v = bd1[t];
        for (int n = 0; n < 128; ++n) v += Wd1[t * 128 + n] * bp[n];
        sbc[t] = v;
        bconstG[t] = v;
    }
    __syncthreads();
    for (int idx = t; idx < 64 * 64; idx += 256) {  // M = Wd1 @ W2eff
        int q = idx >> 6, k = idx & 63;
        float v = 0.0f;
        for (int n = 0; n < 128; ++n) v += Wd1[q * 128 + n] * sW2f[n * 64 + k];
        Mh[idx] = (_Float16)v;
    }
    for (int idx = t; idx < BB * II * 64; idx += 256) {  // Pinit
        int row = idx >> 6, c = idx & 63;
        const float* pc = pos_coords + row * 3;
        Pinit[idx] = pc[0] * sWpd[c * 3 + 0] + pc[1] * sWpd[c * 3 + 1] +
                     pc[2] * sWpd[c * 3 + 2] + sbc[c];
    }
}

// ---------------------------------------------------------------------------
// Encoder: h0 = relu(points @ W1eff.T) -> WMMA: pf = h0@W2eff.T, Q = h0@M.T
// Block = 256 thr = 8 waves; each wave owns a 16-point tile (128 pts/block).
// A fragment (16x32 f16): lane&15 = row M; 16 contiguous K starting at
// k0 = (lane<16)?0:16  -> two ds_load_b128 from row-major LDS [M][64].
// B fragment (32x16 f16): lane&15 = col N; 16 contiguous K starting at
// (lane<16)?0:16 -> two ds_load_b128 from row-major weights [N][64].
// D (16x16 f32): col N = lane&15; VGPR g -> row M = g + ((lane<16)?0:8).
// ---------------------------------------------------------------------------
__device__ __forceinline__ v16h load_a_frag(const _Float16* arow, int hi) {
    const int k0 = hi * 16;
    v8h lo = *(const v8h*)(arow + k0);
    v8h hh = *(const v8h*)(arow + k0 + 8);
    v16h a;
#pragma unroll
    for (int i = 0; i < 8; ++i) { a[i] = lo[i]; a[8 + i] = hh[i]; }
    return a;
}
__device__ __forceinline__ v16h load_b_frag(const _Float16* brow) {
    v8h lo = *(const v8h*)(brow);
    v8h hh = *(const v8h*)(brow + 8);
    v16h b;
#pragma unroll
    for (int i = 0; i < 8; ++i) { b[i] = lo[i]; b[8 + i] = hh[i]; }
    return b;
}

__global__ __launch_bounds__(256)
void encoder_kernel(const float* __restrict__ points,
                    const float* __restrict__ W1eff,
                    const _Float16* __restrict__ W2h,
                    const _Float16* __restrict__ Mh,
                    float* __restrict__ pf_out,  // [B][N][128]
                    float* __restrict__ Q)       // [B][N][64]
{
    __shared__ __align__(16) _Float16 sW2[128 * 64];
    __shared__ __align__(16) _Float16 sM[64 * 64];
    __shared__ __align__(16) _Float16 sA[128 * 64];
    __shared__ float sW1[64 * 3];

    const int b = blockIdx.y;
    const int tile = blockIdx.x * 128;  // point base within batch
    const int t = threadIdx.x;

    for (int idx = t; idx < 128 * 64; idx += 256) sW2[idx] = W2h[idx];
    for (int idx = t; idx < 64 * 64; idx += 256) sM[idx] = Mh[idx];
    if (t < 192) sW1[t] = W1eff[t];
    __syncthreads();

    {   // h0 = relu(points @ W1eff.T): thread -> point t>>1, 32 cols
        const int p = t >> 1;
        const int cb = (t & 1) * 32;
        const float* pp = points + ((size_t)b * NN + tile + p) * 3;
        const float x0 = pp[0], x1 = pp[1], x2 = pp[2];
#pragma unroll
        for (int c = 0; c < 32; ++c) {
            const int cc = cb + c;
            float h = fmaf(x0, sW1[cc * 3 + 0],
                      fmaf(x1, sW1[cc * 3 + 1], x2 * sW1[cc * 3 + 2]));
            sA[p * 64 + cc] = (_Float16)fmaxf(h, 0.0f);
        }
    }
    __syncthreads();

    const int lane = t & 31;
    const int wave = t >> 5;
    const int m = lane & 15;
    const int hi = lane >> 4;

    const _Float16* arow = sA + (wave * 16 + m) * 64;
    const v16h a0 = load_a_frag(arow, hi);        // K = 0..31
    const v16h a1 = load_a_frag(arow + 32, hi);   // K = 32..63

    // pf = h0 @ W2eff.T  (8 n-tiles of 16)
    const size_t prow = ((size_t)b * NN + tile + wave * 16) * EE;
#pragma unroll
    for (int nt = 0; nt < 8; ++nt) {
        const _Float16* brow = sW2 + (nt * 16 + m) * 64 + hi * 16;
        v16h b0 = load_b_frag(brow);
        v16h b1 = load_b_frag(brow + 32);
        v8f acc = {};
        acc = __builtin_amdgcn_wmma_f32_16x16x32_f16(false, a0, false, b0,
                                                     (short)0, acc, false, false);
        acc = __builtin_amdgcn_wmma_f32_16x16x32_f16(false, a1, false, b1,
                                                     (short)0, acc, false, false);
        float* o = pf_out + prow + nt * 16 + m;
#pragma unroll
        for (int g = 0; g < 8; ++g) o[(size_t)(g + hi * 8) * EE] = acc[g];
    }

    // Q = h0 @ M.T  (4 n-tiles of 16)
    const size_t qrow = ((size_t)b * NN + tile + wave * 16) * HH;
#pragma unroll
    for (int nt = 0; nt < 4; ++nt) {
        const _Float16* brow = sM + (nt * 16 + m) * 64 + hi * 16;
        v16h b0 = load_b_frag(brow);
        v16h b1 = load_b_frag(brow + 32);
        v8f acc = {};
        acc = __builtin_amdgcn_wmma_f32_16x16x32_f16(false, a0, false, b0,
                                                     (short)0, acc, false, false);
        acc = __builtin_amdgcn_wmma_f32_16x16x32_f16(false, a1, false, b1,
                                                     (short)0, acc, false, false);
        float* o = Q + qrow + nt * 16 + m;
#pragma unroll
        for (int g = 0; g < 8; ++g) o[(size_t)(g + hi * 8) * HH] = acc[g];
    }
}

// ---------------------------------------------------------------------------
// Initial masks + IoU intersection counts (factored decode; logits never
// materialized). mask[i] bit per thread-point, block-local pair counting,
// integer atomics -> deterministic.
// ---------------------------------------------------------------------------
__global__ __launch_bounds__(256)
void mask_inter_kernel(const float* __restrict__ Q,
                       const float* __restrict__ Pinit,
                       const float* __restrict__ Wd2,
                       const float* __restrict__ bd2,
                       unsigned* __restrict__ inter)
{
    __shared__ float sP[II * HH];
    __shared__ float sw[HH];
    __shared__ unsigned smask[256];
    const int b = blockIdx.y;
    const int n = blockIdx.x * 256 + threadIdx.x;
    const int t = threadIdx.x;

    for (int i = t; i < II * HH; i += 256) sP[i] = Pinit[b * II * HH + i];
    if (t < HH) sw[t] = Wd2[t];
    __syncthreads();

    float qr[HH];
    const float4* q4 = (const float4*)(Q + ((size_t)b * NN + n) * HH);
#pragma unroll
    for (int c = 0; c < HH / 4; ++c) {
        float4 v = q4[c];
        qr[4 * c + 0] = v.x; qr[4 * c + 1] = v.y;
        qr[4 * c + 2] = v.z; qr[4 * c + 3] = v.w;
    }
    const float bias = bd2[0];
    unsigned mword = 0;
    for (int i = 0; i < II; ++i) {
        const float* p = sP + i * HH;
        float s = bias;
#pragma unroll
        for (int c = 0; c < HH; ++c) s += sw[c] * fmaxf(qr[c] + p[c], 0.0f);
        mword |= (s > 0.0f) ? (1u << i) : 0u;   // sigmoid(s)>0.5 <=> s>0
    }
    smask[t] = mword;
    __syncthreads();

    for (int pr = t; pr < II * II; pr += 256) {
        const int i = pr >> 5, j = pr & 31;
        unsigned cnt = 0;
        for (int k = 0; k < 256; ++k)
            cnt += (smask[k] >> i) & (smask[k] >> j) & 1u;
        atomicAdd(&inter[b * II * II + pr], cnt);
    }
}

// ---------------------------------------------------------------------------
// NPC: exact JAX threefry2x32 (key(42) -> (0,42); iota(2048) split halves),
// uniform -> gumbel, first-max argmax over IoU>=0.1 candidates.
// P_ref = P_init - neg @ (Wd1@Wp).T   (bp cancels in pos_embed - neg_embed)
// ---------------------------------------------------------------------------
__device__ __forceinline__ unsigned rotl32(unsigned x, int n) {
    return (x << n) | (x >> (32 - n));
}
__device__ unsigned threefry_elem(unsigned e) {
    const unsigned k0 = 0u, k1 = 42u, k2 = k0 ^ k1 ^ 0x1BD11BDAu;
    unsigned x0, x1; int sel;
    if (e < 1024u) { x0 = e; x1 = e + 1024u; sel = 0; }
    else           { x0 = e - 1024u; x1 = e; sel = 1; }
    x0 += k0; x1 += k1;
    const int RA[4] = {13, 15, 26, 6}, RB[4] = {17, 29, 16, 24};
#define TF_G(R) { for (int r = 0; r < 4; ++r) { x0 += x1; x1 = rotl32(x1, R[r]); x1 ^= x0; } }
    TF_G(RA); x0 += k1; x1 += k2 + 1u;
    TF_G(RB); x0 += k2; x1 += k0 + 2u;
    TF_G(RA); x0 += k0; x1 += k1 + 3u;
    TF_G(RB); x0 += k1; x1 += k2 + 4u;
    TF_G(RA); x0 += k2; x1 += k0 + 5u;
#undef TF_G
    return sel ? x1 : x0;
}
__device__ float gumbel_from_bits(unsigned bits) {
    unsigned fb = (bits >> 9) | 0x3F800000u;
    float u;
    __builtin_memcpy(&u, &fb, 4);
    u -= 1.0f;
    u = u * (1.0f - 1e-20f) + 1e-20f;
    u = fmaxf(u, 1e-20f);
    return -logf(-logf(u));
}

__global__ __launch_bounds__(64)
void npc_kernel(const unsigned* __restrict__ inter,
                const float* __restrict__ pos_coords,
                const float* __restrict__ Wpd,
                const float* __restrict__ Pinit,
                float* __restrict__ Pref)
{
    const int t = threadIdx.x;           // 0..63 -> (b,i)
    const int b = t >> 5, i = t & 31;
    const unsigned* it = inter + b * II * II;
    const float msum_i = (float)it[i * II + i];
    float best = -INFINITY;
    int bidx = -1;
    for (int j = 0; j < II; ++j) {
        if (j == i) continue;
        const float in_ij = (float)it[i * II + j];
        const float un = msum_i + (float)it[j * II + j] - in_ij;
        const float iou = in_ij / (un + 1e-6f);
        if (iou >= 0.1f) {
            const float g = gumbel_from_bits(
                threefry_elem((unsigned)((b * II + i) * II + j)));
            if (g > best) { best = g; bidx = j; }
        }
    }
    float n0 = 0.0f, n1 = 0.0f, n2 = 0.0f;
    if (bidx >= 0) {
        const float* pc = pos_coords + (b * II + bidx) * 3;
        n0 = pc[0]; n1 = pc[1]; n2 = pc[2];
    }
    for (int c = 0; c < HH; ++c) {
        const float adj = n0 * Wpd[c * 3 + 0] + n1 * Wpd[c * 3 + 1] +
                          n2 * Wpd[c * 3 + 2];
        Pref[t * HH + c] = Pinit[t * HH + c] - adj;
    }
}

// ---------------------------------------------------------------------------
// Refined logits: out[b][i][n] = bd2 + sum_c wd2[c]*relu(Q[b,n,c]+Pref[b,i,c])
// ---------------------------------------------------------------------------
__global__ __launch_bounds__(256)
void refined_kernel(const float* __restrict__ Q,
                    const float* __restrict__ Pref,
                    const float* __restrict__ Wd2,
                    const float* __restrict__ bd2,
                    float* __restrict__ out)
{
    __shared__ float sP[II * HH];
    __shared__ float sw[HH];
    const int b = blockIdx.y;
    const int n = blockIdx.x * 256 + threadIdx.x;
    const int t = threadIdx.x;

    for (int i = t; i < II * HH; i += 256) sP[i] = Pref[b * II * HH + i];
    if (t < HH) sw[t] = Wd2[t];
    __syncthreads();

    float qr[HH];
    const float4* q4 = (const float4*)(Q + ((size_t)b * NN + n) * HH);
#pragma unroll
    for (int c = 0; c < HH / 4; ++c) {
        float4 v = q4[c];
        qr[4 * c + 0] = v.x; qr[4 * c + 1] = v.y;
        qr[4 * c + 2] = v.z; qr[4 * c + 3] = v.w;
    }
    const float bias = bd2[0];
    for (int i = 0; i < II; ++i) {
        const float* p = sP + i * HH;
        float s = bias;
#pragma unroll
        for (int c = 0; c < HH; ++c) s += sw[c] * fmaxf(qr[c] + p[c], 0.0f);
        out[((size_t)b * II + i) * NN + n] = s;
    }
}

// ---------------------------------------------------------------------------
extern "C" void kernel_launch(void* const* d_in, const int* in_sizes, int n_in,
                              void* d_out, int out_size, void* d_ws, size_t ws_size,
                              hipStream_t stream) {
    const float* points = (const float*)d_in[0];
    const float* pos    = (const float*)d_in[1];
    const float* W1  = (const float*)d_in[2];
    const float* A1  = (const float*)d_in[3];
    const float* B1  = (const float*)d_in[4];
    const float* W2  = (const float*)d_in[5];
    const float* A2  = (const float*)d_in[6];
    const float* B2  = (const float*)d_in[7];
    const float* Wp  = (const float*)d_in[8];
    const float* bp  = (const float*)d_in[9];
    const float* Wd1 = (const float*)d_in[10];
    const float* bd1 = (const float*)d_in[11];
    const float* Wd2 = (const float*)d_in[12];
    const float* bd2 = (const float*)d_in[13];

    char* w = (char*)d_ws;
    float*     Q      = (float*)(w + WS_Q);
    _Float16*  W2h    = (_Float16*)(w + WS_W2H);
    _Float16*  Mh     = (_Float16*)(w + WS_MH);
    float*     W1eff  = (float*)(w + WS_W1EFF);
    float*     Wpd    = (float*)(w + WS_WPD);
    float*     bconst = (float*)(w + WS_BCONST);
    float*     Pinit  = (float*)(w + WS_PINIT);
    float*     Pref   = (float*)(w + WS_PREF);
    unsigned*  inter  = (unsigned*)(w + WS_INTER);

    float* out_refined = (float*)d_out;                     // [B][I][N]
    float* out_pf      = (float*)d_out + (size_t)BB * II * NN;  // [B][N][E]

    prep_kernel<<<dim3(1), dim3(256), 0, stream>>>(
        W1, A1, B1, W2, A2, B2, Wp, bp, Wd1, bd1, pos,
        W1eff, W2h, Mh, Wpd, bconst, Pinit);

    encoder_kernel<<<dim3(NN / 128, BB), dim3(256), 0, stream>>>(
        points, W1eff, W2h, Mh, out_pf, Q);

    hipMemsetAsync(inter, 0, (size_t)BB * II * II * sizeof(unsigned), stream);

    mask_inter_kernel<<<dim3(NN / 256, BB), dim3(256), 0, stream>>>(
        Q, Pinit, Wd2, bd2, inter);

    npc_kernel<<<dim3(1), dim3(64), 0, stream>>>(inter, pos, Wpd, Pinit, Pref);

    refined_kernel<<<dim3(NN / 256, BB), dim3(256), 0, stream>>>(
        Q, Pref, Wd2, bd2, out_refined);
}
